// Serial_Graphormer_47021301956917
// MI455X (gfx1250) — matrix operand order repaired and checked by reading, not compiled, not run on hardware
//
#include <hip/hip_runtime.h>
#include <hip/hip_bf16.h>
#include <math.h>

typedef __bf16 bf16;
typedef __attribute__((ext_vector_type(16))) __bf16 v16bf;
typedef __attribute__((ext_vector_type(8)))  __bf16 v8bf;
typedef __attribute__((ext_vector_type(4)))  __bf16 v4bf;
typedef __attribute__((ext_vector_type(8)))  float  v8f;
typedef __attribute__((ext_vector_type(4)))  float  v4f;

#define DIMC   128
#define HEADS  4
#define RANKC  32
#define HDC    32
#define BN     2
#define TN     128
#define NN     1024
#define FN     128
#define EDC    512                 // EXPAND*DIM
#define MPB    (NN*FN)             // 131072 columns per batch (n*F+f)
#define ROWS   (BN*TN*NN)          // 262144 per-position rows
#define EPSF   1e-6f
#define XLD    136                 // padded LDS stride for sX (bank spread)
#define HLD    520                 // padded LDS stride for sH

__device__ inline v8f vzero() {
  v8f z;
#pragma unroll
  for (int i = 0; i < 8; ++i) z[i] = 0.0f;
  return z;
}

__device__ inline v8f wmma_bf16(v16bf a, v16bf b, v8f c) {
  return __builtin_amdgcn_wmma_f32_16x16x32_bf16(false, a, false, b, (short)0, c,
                                                 false, false);
}

// ---- A fragment (16x32 bf16), K contiguous in memory, row stride sm elements.
__device__ inline v16bf frag_a_cont(const bf16* s, long sm, int lane) {
  int m = lane & 15, sub = lane >> 4;
  const bf16* p = s + (long)m * sm + 8 * sub;
  v8bf lo = *(const v8bf*)(p);
  v8bf hi = *(const v8bf*)(p + 16);
  return __builtin_shufflevector(lo, hi, 0, 1, 2, 3, 4, 5, 6, 7,
                                 8, 9, 10, 11, 12, 13, 14, 15);
}

// ---- B fragment (32x16 bf16), K contiguous, column stride sn elements.
__device__ inline v16bf frag_b_cont(const bf16* s, long sn, int lane) {
  int n = lane & 15, sub = lane >> 4;
  const bf16* p = s + (long)n * sn + 8 * sub;
  v8bf lo = *(const v8bf*)(p);
  v8bf hi = *(const v8bf*)(p + 16);
  return __builtin_shufflevector(lo, hi, 0, 1, 2, 3, 4, 5, 6, 7,
                                 8, 9, 10, 11, 12, 13, 14, 15);
}

// ---- generic (strided) fragment loaders for the low-rank attention kernels
__device__ inline v16bf frag_a_gen(const bf16* s, long sm, long sk, int lane) {
  int m = lane & 15, sub = lane >> 4;
  v16bf a;
#pragma unroll
  for (int e = 0; e < 16; ++e) {
    int k = (e & 7) + 8 * sub + 16 * (e >> 3);
    a[e] = s[(long)m * sm + (long)k * sk];
  }
  return a;
}
__device__ inline v16bf frag_b_gen(const bf16* s, long sk, long sn, int lane) {
  int n = lane & 15, sub = lane >> 4;
  v16bf b;
#pragma unroll
  for (int e = 0; e < 16; ++e) {
    int k = (e & 7) + 8 * sub + 16 * (e >> 3);
    b[e] = s[(long)k * sk + (long)n * sn];
  }
  return b;
}

// ---- 16x128 GEMM against a 128x128 (o,f) weight: batched B-fragments per
// K-step so loads issue as one clause and WMMAs drain loadcnt incrementally.
__device__ inline void gemm_16x128(const bf16* Abase, const bf16* W, v8f c[8],
                                   int lane) {
#pragma unroll
  for (int kt = 0; kt < 4; ++kt) {
    v16bf a = frag_a_cont(Abase + kt * 32, DIMC, lane);
    v16bf bb[8];
#pragma unroll
    for (int nt = 0; nt < 8; ++nt)
      bb[nt] = frag_b_cont(W + (nt * 16) * DIMC + kt * 32, DIMC, lane);
#pragma unroll
    for (int nt = 0; nt < 8; ++nt) c[nt] = wmma_bf16(a, bb[nt], c[nt]);
  }
}

// ---------------------------------------------------------------------------
// Kernel 0: weight prep. bf16 copies of q_w, v_w, w1, w2, w3; build 128x128
// block-diagonal key matrix Kb[h*32+r][h*32+d] = key_p[r,h,d] / sqrt(32).
// ---------------------------------------------------------------------------
__global__ void k_prep(const float* __restrict__ qw, const float* __restrict__ vw,
                       const float* __restrict__ keyp,
                       const float* __restrict__ w1, const float* __restrict__ w2,
                       const float* __restrict__ w3,
                       bf16* qwb, bf16* vwb, bf16* kb,
                       bf16* w1b, bf16* w2b, bf16* w3b) {
  int i = blockIdx.x * blockDim.x + threadIdx.x;
  if (i < 16384) {
    qwb[i] = (bf16)qw[i];
  } else if (i < 32768) {
    int j = i - 16384; vwb[j] = (bf16)vw[j];
  } else if (i < 49152) {
    int j = i - 32768;
    int o = j >> 7, f = j & 127;
    int h = o >> 5, r = o & 31;
    float v = ((f >> 5) == h) ? keyp[r * 128 + f] * 0.17677669529663687f : 0.0f;
    kb[j] = (bf16)v;
  } else if (i < 114688) {
    int j = i - 49152; w1b[j] = (bf16)w1[j];
  } else if (i < 180224) {
    int j = i - 114688; w2b[j] = (bf16)w2[j];
  } else if (i < 245760) {
    int j = i - 180224; w3b[j] = (bf16)w3[j];
  }
}

// ---------------------------------------------------------------------------
// Kernel A: rmsnorm over the t axis, 4 columns per thread (float4 in, bf16x4 out).
// ---------------------------------------------------------------------------
__global__ void k_rms1(const float* __restrict__ x, const float* __restrict__ norm_w,
                       bf16* __restrict__ x1) {
  int g = blockIdx.x * 256 + threadIdx.x;   // 0..65535 column groups of 4
  int b = g >> 15;
  int m = (g & 32767) * 4;
  const float* col = x + (size_t)b * TN * MPB + m;
  v4f ss;
#pragma unroll
  for (int j = 0; j < 4; ++j) ss[j] = 0.0f;
  for (int t = 0; t < TN; ++t) {
    v4f v = *(const v4f*)(col + (size_t)t * MPB);
    ss += v * v;
  }
  v4f inv;
#pragma unroll
  for (int j = 0; j < 4; ++j)
    inv[j] = rsqrtf(ss[j] * (1.0f / TN) + EPSF) * norm_w[(m + j) & 127];
  bf16* o = x1 + (size_t)b * TN * MPB + m;
  for (int t = 0; t < TN; ++t) {
    v4f v = *(const v4f*)(col + (size_t)t * MPB);
    v4bf w;
#pragma unroll
    for (int j = 0; j < 4; ++j) w[j] = (bf16)(v[j] * inv[j]);
    *(v4bf*)(o + (size_t)t * MPB) = w;
  }
}

// ---------------------------------------------------------------------------
// Kernel B: fused  q = x1 qW^T + qb ; attn = q Kb^T ; rank softmax ; xv.
// ---------------------------------------------------------------------------
__global__ __launch_bounds__(256) void k_qkv(
    const bf16* __restrict__ x1, const bf16* __restrict__ qwb,
    const bf16* __restrict__ vwb, const bf16* __restrict__ kb,
    const float* __restrict__ qb_, const float* __restrict__ vb_,
    float* __restrict__ attn_g, bf16* __restrict__ smr_g, bf16* __restrict__ xv_g) {
  __shared__ float lds[8 * 2048];   // 8KB/wave: q (bf16 16x128) then attn (f32 16x128)
  int lane = threadIdx.x & 31, wv = threadIdx.x >> 5, sub = lane >> 4;
  size_t rowbase = ((size_t)blockIdx.x * 8 + wv) * 16;
  float* L = lds + wv * 2048;
  bf16* qL = (bf16*)L;
  const bf16* A0 = x1 + rowbase * DIMC;

  // ---- q GEMM (16x128x128)
  {
    v8f c[8];
#pragma unroll
    for (int nt = 0; nt < 8; ++nt) c[nt] = vzero();
    gemm_16x128(A0, qwb, c, lane);
#pragma unroll
    for (int nt = 0; nt < 8; ++nt) {
      int co = nt * 16 + (lane & 15);
      float bias = qb_[co];
#pragma unroll
      for (int i = 0; i < 8; ++i)
        qL[(i + 8 * sub) * DIMC + co] = (bf16)(c[nt][i] + bias);
    }
  }
  // ---- attn GEMM (16x128x128, block-diagonal key weights, scale folded in)
  {
    v8f c[8];
#pragma unroll
    for (int nt = 0; nt < 8; ++nt) c[nt] = vzero();
    gemm_16x128(qL, kb, c, lane);
#pragma unroll
    for (int nt = 0; nt < 8; ++nt) {
#pragma unroll
      for (int i = 0; i < 8; ++i)
        L[(i + 8 * sub) * DIMC + nt * 16 + (lane & 15)] = c[nt][i];
    }
  }
  // ---- dump logits (f32, float4-vectorized) + rank softmax -> smr (bf16)
  for (int q4 = lane; q4 < 512; q4 += 32)
    *(v4f*)(attn_g + rowbase * DIMC + q4 * 4) = *(const v4f*)(L + q4 * 4);
  for (int tsk = lane; tsk < 64; tsk += 32) {
    int row = tsk >> 2, ch = tsk & 3;
    const float* p = L + row * DIMC + ch * 32;
    float mx = -1e30f;
    for (int j = 0; j < 32; ++j) mx = fmaxf(mx, p[j]);
    float s = 0.0f;
    for (int j = 0; j < 32; ++j) s += __expf(p[j] - mx);
    float is = 1.0f / s;
    bf16* o = smr_g + (rowbase + row) * DIMC + ch * 32;
    for (int j = 0; j < 32; ++j) o[j] = (bf16)(__expf(p[j] - mx) * is);
  }
  // ---- xv GEMM (16x128x128)
  {
    v8f c[8];
#pragma unroll
    for (int nt = 0; nt < 8; ++nt) c[nt] = vzero();
    gemm_16x128(A0, vwb, c, lane);
#pragma unroll
    for (int nt = 0; nt < 8; ++nt) {
      int co = nt * 16 + (lane & 15);
      float bias = vb_[co];
#pragma unroll
      for (int i = 0; i < 8; ++i)
        xv_g[(rowbase + i + 8 * sub) * DIMC + co] = (bf16)(c[nt][i] + bias);
    }
  }
}

// ---------------------------------------------------------------------------
// Kernel C1: node-axis softmax stats, 4 (h,r) columns per thread (float4).
// ---------------------------------------------------------------------------
__global__ void k_colstats(const float* __restrict__ attn_g,
                           float* __restrict__ cmax, float* __restrict__ csum) {
  int g = blockIdx.x * 256 + threadIdx.x;    // 0..8191 groups of 4 columns
  int bt = g >> 5;                            // 32 groups per (b*T+t)
  int hr = (g & 31) * 4;
  const float* base = attn_g + (size_t)bt * NN * DIMC + hr;
  v4f mx;
#pragma unroll
  for (int j = 0; j < 4; ++j) mx[j] = -1e30f;
  for (int n = 0; n < NN; ++n) {
    v4f v = *(const v4f*)(base + (size_t)n * DIMC);
#pragma unroll
    for (int j = 0; j < 4; ++j) mx[j] = fmaxf(mx[j], v[j]);
  }
  v4f s;
#pragma unroll
  for (int j = 0; j < 4; ++j) s[j] = 0.0f;
  for (int n = 0; n < NN; ++n) {
    v4f v = *(const v4f*)(base + (size_t)n * DIMC);
#pragma unroll
    for (int j = 0; j < 4; ++j) s[j] += __expf(v[j] - mx[j]);
  }
  *(v4f*)(cmax + g * 4) = mx;
  *(v4f*)(csum + g * 4) = s;
}

// ---------------------------------------------------------------------------
// Kernel C2: landmark aggregation  v[r,d] = sum_n smr[n,r] * xv[n,d]
// One wave per (b,t,h); K=1024 with depth-2 software pipelining.
// ---------------------------------------------------------------------------
__global__ __launch_bounds__(256) void k_land(const bf16* __restrict__ smr,
                                              const bf16* __restrict__ xv,
                                              float* __restrict__ vland) {
  int lane = threadIdx.x & 31, wv = threadIdx.x >> 5, sub = lane >> 4;
  int grp = blockIdx.x * 8 + wv;   // (b*T+t)*4 + h
  int bt = grp >> 2, h = grp & 3;
  const bf16* sbase = smr + (size_t)bt * NN * DIMC + h * 32;
  const bf16* vbase = xv + (size_t)bt * NN * DIMC + h * 32;
  v8f c[2][2];
#pragma unroll
  for (int mt = 0; mt < 2; ++mt)
#pragma unroll
    for (int nt = 0; nt < 2; ++nt) c[mt][nt] = vzero();
  v16bf a[2], b[2];
#pragma unroll
  for (int mt = 0; mt < 2; ++mt)
    a[mt] = frag_a_gen(sbase + mt * 16, 1, DIMC, lane);
#pragma unroll
  for (int nt = 0; nt < 2; ++nt)
    b[nt] = frag_b_gen(vbase + nt * 16, DIMC, 1, lane);
  for (int kt = 0; kt < 32; ++kt) {
    v16bf an[2] = {a[0], a[1]}, bn[2] = {b[0], b[1]};
    if (kt < 31) {
      long n1 = (long)(kt + 1) * 32;
#pragma unroll
      for (int mt = 0; mt < 2; ++mt)
        an[mt] = frag_a_gen(sbase + n1 * DIMC + mt * 16, 1, DIMC, lane);
#pragma unroll
      for (int nt = 0; nt < 2; ++nt)
        bn[nt] = frag_b_gen(vbase + n1 * DIMC + nt * 16, DIMC, 1, lane);
    }
#pragma unroll
    for (int mt = 0; mt < 2; ++mt)
#pragma unroll
      for (int nt = 0; nt < 2; ++nt) c[mt][nt] = wmma_bf16(a[mt], b[nt], c[mt][nt]);
#pragma unroll
    for (int mt = 0; mt < 2; ++mt) { a[mt] = an[mt]; b[mt] = bn[mt]; }
  }
  float* o = vland + (size_t)grp * 1024;
#pragma unroll
  for (int mt = 0; mt < 2; ++mt)
#pragma unroll
    for (int nt = 0; nt < 2; ++nt)
#pragma unroll
      for (int i = 0; i < 8; ++i) {
        int r = mt * 16 + i + 8 * sub;
        int d = nt * 16 + (lane & 15);
        o[r * 32 + d] = c[mt][nt][i];
      }
}

// ---------------------------------------------------------------------------
// Kernel D: v2 = softmax_n(attn) @ vland ; x2 = sig(alpha)*xv + sig(beta)*v2 + x.
// ---------------------------------------------------------------------------
__global__ __launch_bounds__(256) void k_apply(
    const float* __restrict__ attn_g, const float* __restrict__ cmax,
    const float* __restrict__ csum, const float* __restrict__ vland,
    const bf16* __restrict__ xv, const float* __restrict__ x,
    const float* __restrict__ alpha, const float* __restrict__ beta,
    float* __restrict__ x2) {
  __shared__ float smx[8][32];
  __shared__ float srs[8][32];
  __shared__ bf16 svl[8][1024];
  int lane = threadIdx.x & 31, wv = threadIdx.x >> 5, sub = lane >> 4;
  int grp = blockIdx.x * 8 + wv;
  int bt = grp >> 2, h = grp & 3;
  smx[wv][lane] = cmax[grp * 32 + lane];
  srs[wv][lane] = 1.0f / csum[grp * 32 + lane];
  for (int j = lane; j < 1024; j += 32)
    svl[wv][j] = (bf16)vland[(size_t)grp * 1024 + j];
  float sa = 1.0f / (1.0f + __expf(-alpha[h]));
  float sb = 1.0f / (1.0f + __expf(-beta[h]));
  const float* abase = attn_g + (size_t)bt * NN * DIMC + h * 32;
  const bf16* xvb = xv + (size_t)bt * NN * DIMC + h * 32;
  const float* xb = x + (size_t)bt * NN * DIMC + h * 32;
  float* ob = x2 + (size_t)bt * NN * DIMC + h * 32;
  v16bf b[2];
  b[0] = frag_b_gen(&svl[wv][0], 32, 1, lane);
  b[1] = frag_b_gen(&svl[wv][16], 32, 1, lane);
  int m = lane & 15;
  for (int mt = 0; mt < 64; ++mt) {
    long n0 = (long)mt * 16;
    const float* p = abase + (n0 + m) * DIMC + 8 * sub;
    v8f lo = *(const v8f*)(p);          // r = 8*sub + 0..7
    v8f hi = *(const v8f*)(p + 16);     // r = 8*sub + 16..23
    v16bf a;
#pragma unroll
    for (int e = 0; e < 8; ++e) {
      int rl = e + 8 * sub;
      int rh = rl + 16;
      a[e]     = (bf16)(__expf(lo[e] - smx[wv][rl]) * srs[wv][rl]);
      a[e + 8] = (bf16)(__expf(hi[e] - smx[wv][rh]) * srs[wv][rh]);
    }
#pragma unroll
    for (int nt = 0; nt < 2; ++nt) {
      v8f c = wmma_bf16(a, b[nt], vzero());
#pragma unroll
      for (int i = 0; i < 8; ++i) {
        long n = n0 + i + 8 * sub;
        int d = nt * 16 + (lane & 15);
        size_t idx = (size_t)n * DIMC + d;
        ob[idx] = sa * (float)xvb[idx] + sb * c[i] + xb[idx];
      }
    }
  }
}

// ---------------------------------------------------------------------------
// Kernel E: fully fused FFN. x2 tile (128t x 64m) -> LDS, rmsnorm over t,
// H = silu(W1 X + b1)*(W2 X + b2) kept bf16 in LDS (transposed, padded),
// then out = W3 H + b3 + x2. All fragment batches preloaded / double-buffered.
// ---------------------------------------------------------------------------
__global__ __launch_bounds__(256) void k_ffn(
    const float* __restrict__ x2, const float* __restrict__ norm_w,
    const bf16* __restrict__ w1b, const bf16* __restrict__ w2b,
    const bf16* __restrict__ w3b, const float* __restrict__ b1,
    const float* __restrict__ b2, const float* __restrict__ b3,
    float* __restrict__ out) {
  __shared__ float sx2[128 * 64];    // residual + norm source (32 KB), [t][m]
  __shared__ bf16 sX[64 * XLD];      // normalized activations, [m][t] (17 KB)
  __shared__ bf16 sH[64 * HLD];      // hidden, [m][o] (65 KB)
  __shared__ float psum[256];
  __shared__ float sinv[64];
  int tid = threadIdx.x;
  int bb_ = blockIdx.x >> 11;              // batch
  int m0 = (blockIdx.x & 2047) * 64;       // column block within (n,f) space
  const float* xbase = x2 + (size_t)bb_ * TN * MPB + m0;
  float ss = 0.0f;
  for (int idx = tid; idx < 8192; idx += 256) {   // col = idx&63 fixed per thread
    int t = idx >> 6, c = idx & 63;
    float v = xbase[(size_t)t * MPB + c];
    sx2[idx] = v;
    ss += v * v;
  }
  psum[tid] = ss;
  __syncthreads();
  if (tid < 64) {
    float s = psum[tid] + psum[tid + 64] + psum[tid + 128] + psum[tid + 192];
    sinv[tid] = rsqrtf(s * (1.0f / TN) + EPSF) * norm_w[(m0 + tid) & 127];
  }
  __syncthreads();
  for (int idx = tid; idx < 8192; idx += 256) {
    int t = idx >> 6, c = idx & 63;
    sX[c * XLD + t] = (bf16)(sx2[idx] * sinv[c]);   // transposed: K contiguous
  }
  __syncthreads();

  int lane = tid & 31, wv = tid >> 5, sub = lane >> 4;
  __builtin_prefetch(w3b + wv * 8192, 0, 1);   // warm W3 for GEMM3
  // GEMM1/GEMM2 + silu*mul -> sH   (512x64, K=128), fragments batch-preloaded
  for (int j = 0; j < 16; ++j) {
    int tt = wv * 16 + j;
    int ot = tt >> 2, mt = tt & 3;
    v16bf xbv[4], a1v[4], a2v[4];
#pragma unroll
    for (int kt = 0; kt < 4; ++kt)
      xbv[kt] = frag_b_cont(sX + (mt * 16) * XLD + kt * 32, XLD, lane);
#pragma unroll
    for (int kt = 0; kt < 4; ++kt)
      a1v[kt] = frag_a_cont(w1b + (ot * 16) * 128 + kt * 32, 128, lane);
#pragma unroll
    for (int kt = 0; kt < 4; ++kt)
      a2v[kt] = frag_a_cont(w2b + (ot * 16) * 128 + kt * 32, 128, lane);
    v8f c1 = vzero(), c2 = vzero();
#pragma unroll
    for (int kt = 0; kt < 4; ++kt) {
      c1 = wmma_bf16(a1v[kt], xbv[kt], c1);
      c2 = wmma_bf16(a2v[kt], xbv[kt], c2);
    }
    v8bf hv8;
#pragma unroll
    for (int i = 0; i < 8; ++i) {
      int o = ot * 16 + i + 8 * sub;
      float h1 = c1[i] + b1[o];
      float h2 = c2[i] + b2[o];
      hv8[i] = (bf16)((h1 / (1.0f + __expf(-h1))) * h2);   // silu(h1)*h2
    }
    *(v8bf*)(sH + (mt * 16 + (lane & 15)) * HLD + ot * 16 + 8 * sub) = hv8;
  }
  __syncthreads();
  // GEMM3: out = W3 H + b3 + residual   (128x64, K=512), depth-2 pipelined
  for (int j = 0; j < 4; ++j) {
    int tt = wv * 4 + j;
    int ot = tt >> 2, mt = tt & 3;
    const bf16* Aw = w3b + (ot * 16) * 512;
    const bf16* Bh = sH + (mt * 16) * HLD;
    v8f c = vzero();
    v16bf aC = frag_a_cont(Aw, 512, lane);
    v16bf bC = frag_b_cont(Bh, HLD, lane);
#pragma unroll
    for (int kt = 0; kt < 16; ++kt) {
      v16bf aN = aC, bN = bC;
      if (kt < 15) {
        aN = frag_a_cont(Aw + (kt + 1) * 32, 512, lane);
        bN = frag_b_cont(Bh + (kt + 1) * 32, HLD, lane);
      }
      c = wmma_bf16(aC, bC, c);
      aC = aN;
      bC = bN;
    }
#pragma unroll
    for (int i = 0; i < 8; ++i) {
      int trow = ot * 16 + i + 8 * sub;
      int mc = mt * 16 + (lane & 15);
      out[(size_t)(bb_ * TN + trow) * MPB + m0 + mc] =
          c[i] + b3[trow] + sx2[trow * 64 + mc];
    }
  }
}

// ---------------------------------------------------------------------------
extern "C" void kernel_launch(void* const* d_in, const int* in_sizes, int n_in,
                              void* d_out, int out_size, void* d_ws, size_t ws_size,
                              hipStream_t stream) {
  const float* x      = (const float*)d_in[0];
  const float* norm_w = (const float*)d_in[1];
  const float* q_w    = (const float*)d_in[2];
  const float* q_b    = (const float*)d_in[3];
  const float* key_p  = (const float*)d_in[4];
  const float* v_w    = (const float*)d_in[5];
  const float* v_b    = (const float*)d_in[6];
  const float* alpha  = (const float*)d_in[7];
  const float* beta   = (const float*)d_in[8];
  const float* w1     = (const float*)d_in[9];
  const float* b1     = (const float*)d_in[10];
  const float* w2     = (const float*)d_in[11];
  const float* b2     = (const float*)d_in[12];
  const float* w3     = (const float*)d_in[13];
  const float* b3     = (const float*)d_in[14];
  float* out = (float*)d_out;
  char* ws = (char*)d_ws;

  bf16*  qwb   = (bf16*)(ws + 0);            //  32 KB
  bf16*  vwb   = (bf16*)(ws + 32768);        //  32 KB
  bf16*  kb    = (bf16*)(ws + 65536);        //  32 KB
  bf16*  w1b   = (bf16*)(ws + 98304);        // 128 KB
  bf16*  w2b   = (bf16*)(ws + 229376);       // 128 KB
  bf16*  w3b   = (bf16*)(ws + 360448);       // 128 KB
  bf16*  x1    = (bf16*)(ws + 491520);       //  64 MB
  float* attn  = (float*)(ws + 67600384);    // 128 MB (f32 logits)
  bf16*  smr   = (bf16*)(ws + 201818112);    //  64 MB
  bf16*  xv    = (bf16*)(ws + 268926976);    //  64 MB
  float* vland = (float*)(ws + 336035840);   //  16 MB
  float* cmax  = (float*)(ws + 352813056);   // 128 KB
  float* csum  = (float*)(ws + 352944128);   // 128 KB
  float* x2    = (float*)(ws + 353075200);   // 128 MB

  k_prep<<<960, 256, 0, stream>>>(q_w, v_w, key_p, w1, w2, w3,
                                  qwb, vwb, kb, w1b, w2b, w3b);
  k_rms1<<<256, 256, 0, stream>>>(x, norm_w, x1);
  k_qkv<<<2048, 256, 0, stream>>>(x1, qwb, vwb, kb, q_b, v_b, attn, smr, xv);
  k_colstats<<<32, 256, 0, stream>>>(attn, cmax, csum);
  k_land<<<128, 256, 0, stream>>>(smr, xv, vland);
  k_apply<<<128, 256, 0, stream>>>(attn, cmax, csum, vland, xv, x, alpha, beta, x2);
  k_ffn<<<4096, 256, 0, stream>>>(x2, norm_w, w1b, w2b, w3b, b1, b2, b3, out);
}
